// MemoryEfficientIntraViewAttention_55173149884509
// MI455X (gfx1250) — matrix operand order repaired and checked by reading, not compiled, not run on hardware
//
#include <hip/hip_runtime.h>
#include <hip/hip_bf16.h>

typedef __attribute__((ext_vector_type(16))) _Float16 v16h;
typedef __attribute__((ext_vector_type(8)))  _Float16 v8h;
typedef __attribute__((ext_vector_type(8)))  float    v8f;
typedef __attribute__((ext_vector_type(4)))  int      v4i;

// Problem constants
constexpr int N_TOK = 12288;   // tokens
constexpr int D     = 512;     // model dim == inner dim
constexpr int NH    = 8;       // heads
constexpr int DH    = 64;      // dim_head
constexpr int WMAP  = 384;     // feature-map row length (12*H)
constexpr int LWIN  = 2048;    // tokens per window (H * Ww = 32*64)
constexpr int LDP   = 72;      // padded LDS row pitch (halves) -> conflict-free frags

// ---------------------------------------------------------------------------
// CDNA5 async-to-LDS support (guarded; falls back to load + ds_store staging)
// ---------------------------------------------------------------------------
#define GAS __attribute__((address_space(1)))
#define LAS __attribute__((address_space(3)))

#if defined(__gfx1250__) && __has_builtin(__builtin_amdgcn_global_load_async_to_lds_b128)
#define HAVE_ASYNC_LDS 1
// builtin wants (int4 AS1*, int4 AS3*, imm offset, imm cpol)
#define GPTR(p) (GAS v4i*)(GAS void*)(void*)(p)
#define LPTR(p) (LAS v4i*)(LAS void*)(void*)(p)
#else
#define HAVE_ASYNC_LDS 0
#endif

#if defined(__gfx1250__) && __has_builtin(__builtin_amdgcn_s_wait_asynccnt)
#define WAIT_ASYNC(n) __builtin_amdgcn_s_wait_asynccnt(n)
#elif defined(__gfx1250__)
#define WAIT_ASYNC(n) asm volatile("s_wait_asynccnt %0" ::"i"(n) : "memory")
#else
#define WAIT_ASYNC(n)
#endif

// ---------------------------------------------------------------------------
// WMMA helpers (f16 16x16x32, f32 accum). Layouts per CDNA5 ISA 7.12.2.
// ---------------------------------------------------------------------------
__device__ __forceinline__ v16h make_frag(const _Float16* p0, const _Float16* p1) {
  v8h lo = *(const v8h*)p0;
  v8h hi = *(const v8h*)p1;
  v16h f;
#pragma unroll
  for (int i = 0; i < 8; ++i) { f[i] = lo[i]; f[i + 8] = hi[i]; }
  return f;
}

// A fragment: 16x32 tile of row-major matrix, base at (row0, k0), ld in halves.
__device__ __forceinline__ v16h load_a(const _Float16* base, int ld, int lane) {
  const int grp = lane >> 4, r = lane & 15;
  const _Float16* p = base + r * ld + grp * 8;
  return make_frag(p, p + 16);
}

// B fragment from B-transposed (n-major) storage: base at (n0 row, k0), ld in halves.
__device__ __forceinline__ v16h load_b(const _Float16* base, int ld, int lane) {
  const int grp = lane >> 4, n = lane & 15;
  const _Float16* p = base + n * ld + grp * 8;
  return make_frag(p, p + 16);
}

__device__ __forceinline__ v8f wmma16(v16h a, v16h b, v8f c) {
  return __builtin_amdgcn_wmma_f32_16x16x32_f16(false, a, false, b, (short)0, c,
                                                false, false);
}

// ---------------------------------------------------------------------------
// fp32 -> f16 conversion (x and weights), 8 elems/thread
// ---------------------------------------------------------------------------
__global__ void cvt_kernel(const float* __restrict__ src, _Float16* __restrict__ dst,
                           int n8) {
  int i = blockIdx.x * blockDim.x + threadIdx.x;
  if (i < n8) {
    const float4* s = (const float4*)(src + i * 8);
    float4 a = s[0], b = s[1];
    v8h o;
    o[0] = (_Float16)a.x; o[1] = (_Float16)a.y; o[2] = (_Float16)a.z; o[3] = (_Float16)a.w;
    o[4] = (_Float16)b.x; o[5] = (_Float16)b.y; o[6] = (_Float16)b.z; o[7] = (_Float16)b.w;
    *(v8h*)(dst + i * 8) = o;
  }
}

// ---------------------------------------------------------------------------
// QKV projection: y = x @ W^T, written directly into windowed layout.
//   z=0: Q -> Qw[g][h][l][d]  (pre-scaled by 1/sqrt(64))
//   z=1: K -> Kw[g][h][l][d]
//   z=2: V -> Vt[g][h][d][l]  (transposed for the P@V B-fragments)
// Wave computes 32x64 (two A-frags share each B-frag). grid(48,8,3) block(32,8).
// ---------------------------------------------------------------------------
__global__ __launch_bounds__(256)
void qkv_kernel(const _Float16* __restrict__ xh,
                const _Float16* __restrict__ Wqh,
                const _Float16* __restrict__ Wkh,
                const _Float16* __restrict__ Wvh,
                _Float16* __restrict__ Qw, _Float16* __restrict__ Kw,
                _Float16* __restrict__ Vt) {
  const int lane = threadIdx.x;
  const int z    = blockIdx.z;
  const int head = blockIdx.y;
  const int m0   = (blockIdx.x * 8 + threadIdx.y) * 32;   // token tile base
  const int grp  = lane >> 4, lr = lane & 15;

  const _Float16* W = (z == 0) ? Wqh : (z == 1) ? Wkh : Wvh;

  v8f acc[2][4] = {};
#pragma unroll
  for (int ks = 0; ks < 16; ++ks) {
    const int k0 = ks * 32;
    v16h a0 = load_a(xh + (size_t)m0 * D + k0, D, lane);
    v16h a1 = load_a(xh + (size_t)(m0 + 16) * D + k0, D, lane);
#pragma unroll
    for (int t = 0; t < 4; ++t) {
      v16h b = load_b(W + (head * DH + t * 16) * D + k0, D, lane);
      acc[0][t] = wmma16(a0, b, acc[0][t]);
      acc[1][t] = wmma16(a1, b, acc[1][t]);
    }
  }

  // token -> (window g, in-window l); 32-row tiles never cross window chunks
  const int hrow  = m0 / WMAP;
  const int rem   = m0 % WMAP;
  const int g     = rem >> 6;
  const int lbase = hrow * 64 + (rem & 63);
  const float scale = (z == 0) ? 0.125f : 1.0f;

  if (z < 2) {
    _Float16* P = (z == 0 ? Qw : Kw) + (size_t)(g * NH + head) * LWIN * DH;
#pragma unroll
    for (int mi = 0; mi < 2; ++mi) {
#pragma unroll
      for (int t = 0; t < 4; ++t) {
        const int d = t * 16 + lr;
#pragma unroll
        for (int r = 0; r < 8; ++r) {
          const int l = lbase + mi * 16 + grp * 8 + r;    // C row = r + 8*grp
          P[l * DH + d] = (_Float16)(acc[mi][t][r] * scale);
        }
      }
    }
  } else {
    _Float16* P = Vt + (size_t)(g * NH + head) * DH * LWIN;
#pragma unroll
    for (int mi = 0; mi < 2; ++mi) {
#pragma unroll
      for (int t = 0; t < 4; ++t) {
        const int d = t * 16 + lr;
        v8h pk;
#pragma unroll
        for (int r = 0; r < 8; ++r) pk[r] = (_Float16)acc[mi][t][r];
        *(v8h*)(P + d * LWIN + lbase + mi * 16 + grp * 8) = pk;
      }
    }
  }
}

// ---------------------------------------------------------------------------
// Flash attention per (window, head). Block = 8 waves x 16 query rows = 128
// rows. K/V 64-key tiles are staged once per block into double-buffered LDS
// via async-to-LDS DMA (ASYNCcnt) and consumed as ds-fragment reads by all
// waves. grid(16,8,6) block(32,8).
// ---------------------------------------------------------------------------
__global__ __launch_bounds__(256)
void attn_kernel(const _Float16* __restrict__ Qw,
                 const _Float16* __restrict__ Kw,
                 const _Float16* __restrict__ Vt,
                 _Float16* __restrict__ Ow) {
  __shared__ __align__(16) _Float16 Kt[2][64 * LDP];   // key-major, padded
  __shared__ __align__(16) _Float16 Vs[2][64 * LDP];   // d-major,   padded
  __shared__ __align__(16) _Float16 Pt[8][16 * LDP];   // per-wave P transpose

  const int lane = threadIdx.x, wv = threadIdx.y;
  const int tid  = wv * 32 + lane;
  const int g = blockIdx.z, head = blockIdx.y;
  const int q0 = blockIdx.x * 128 + wv * 16;
  const int grp = lane >> 4, lr = lane & 15;

  const _Float16* Q = Qw + (size_t)(g * NH + head) * LWIN * DH;
  const _Float16* K = Kw + (size_t)(g * NH + head) * LWIN * DH;
  const _Float16* V = Vt + (size_t)(g * NH + head) * DH * LWIN;

  // this thread's two staging chunks (8 halves each) per 64x64 tile
  const int row0 = tid >> 3,          col0 = (tid & 7) * 8;         // chunks 0..255
  const int row1 = (tid + 256) >> 3,  col1 = col0;                  // chunks 256..511

  auto stage = [&](int kb, int buf) {
    const int key0 = kb * 64;
    const _Float16* gK = K + (size_t)key0 * DH;   // [key][d]
    const _Float16* gV = V + key0;                // [d][key]
#if HAVE_ASYNC_LDS
    __builtin_amdgcn_global_load_async_to_lds_b128(
        GPTR(gK + row0 * DH + col0), LPTR(&Kt[buf][row0 * LDP + col0]), 0, 0);
    __builtin_amdgcn_global_load_async_to_lds_b128(
        GPTR(gK + row1 * DH + col1), LPTR(&Kt[buf][row1 * LDP + col1]), 0, 0);
    __builtin_amdgcn_global_load_async_to_lds_b128(
        GPTR(gV + (size_t)row0 * LWIN + col0), LPTR(&Vs[buf][row0 * LDP + col0]), 0, 0);
    __builtin_amdgcn_global_load_async_to_lds_b128(
        GPTR(gV + (size_t)row1 * LWIN + col1), LPTR(&Vs[buf][row1 * LDP + col1]), 0, 0);
#else
    *(v8h*)(&Kt[buf][row0 * LDP + col0]) = *(const v8h*)(gK + row0 * DH + col0);
    *(v8h*)(&Kt[buf][row1 * LDP + col1]) = *(const v8h*)(gK + row1 * DH + col1);
    *(v8h*)(&Vs[buf][row0 * LDP + col0]) = *(const v8h*)(gV + (size_t)row0 * LWIN + col0);
    *(v8h*)(&Vs[buf][row1 * LDP + col1]) = *(const v8h*)(gV + (size_t)row1 * LWIN + col1);
#endif
  };

  // Q A-fragments resident for the whole key loop
  v16h aq[2];
#pragma unroll
  for (int kk = 0; kk < 2; ++kk) aq[kk] = load_a(Q + q0 * DH + kk * 32, DH, lane);

  float mrow[8], lrow[8];
#pragma unroll
  for (int r = 0; r < 8; ++r) { mrow[r] = -1e30f; lrow[r] = 0.0f; }
  v8f o[4] = {};

  _Float16* pbuf = &Pt[wv][0];
  constexpr int NT = LWIN / 64;

  stage(0, 0);                                   // prefetch tile 0

  for (int kb = 0; kb < NT; ++kb) {
    const int cur = kb & 1;
    if (kb + 1 < NT) {
      stage(kb + 1, cur ^ 1);                    // overlap DMA of next tile
      WAIT_ASYNC(4);                             // tile kb's 4 chunks landed
    } else {
      WAIT_ASYNC(0);
    }
    __syncthreads();                             // tile kb visible to all waves

    const _Float16* Kb = &Kt[cur][0];
    const _Float16* Vb = &Vs[cur][0];

    // S = Q @ K^T  (Q pre-scaled)
    v8f s[4] = {};
#pragma unroll
    for (int t = 0; t < 4; ++t) {
#pragma unroll
      for (int kk = 0; kk < 2; ++kk) {
        v16h b = load_b(Kb + (t * 16) * LDP + kk * 32, LDP, lane);
        s[t] = wmma16(aq[kk], b, s[t]);
      }
    }

    // online softmax in C layout: row = r + 8*grp, cols over 16 lanes x 4 tiles
    float scl[8];
#pragma unroll
    for (int r = 0; r < 8; ++r) {
      float mx = fmaxf(fmaxf(s[0][r], s[1][r]), fmaxf(s[2][r], s[3][r]));
#pragma unroll
      for (int off = 8; off >= 1; off >>= 1)
        mx = fmaxf(mx, __shfl_xor(mx, off, 32));
      const float mn = fmaxf(mrow[r], mx);
      scl[r] = __expf(mrow[r] - mn);
      mrow[r] = mn;
    }
    float rs[8];
#pragma unroll
    for (int r = 0; r < 8; ++r) rs[r] = 0.0f;
#pragma unroll
    for (int t = 0; t < 4; ++t) {
#pragma unroll
      for (int r = 0; r < 8; ++r) {
        const float p = __expf(s[t][r] - mrow[r]);
        s[t][r] = p;
        rs[r] += p;
      }
    }
#pragma unroll
    for (int r = 0; r < 8; ++r) {
#pragma unroll
      for (int off = 8; off >= 1; off >>= 1)
        rs[r] += __shfl_xor(rs[r], off, 32);
      lrow[r] = lrow[r] * scl[r] + rs[r];
    }
#pragma unroll
    for (int t = 0; t < 4; ++t)
#pragma unroll
      for (int r = 0; r < 8; ++r) o[t][r] *= scl[r];

    // probabilities: C layout -> LDS -> A-fragment layout (wave-local)
#pragma unroll
    for (int t = 0; t < 4; ++t)
#pragma unroll
      for (int r = 0; r < 8; ++r)
        pbuf[(grp * 8 + r) * LDP + t * 16 + lr] = (_Float16)s[t][r];
    __builtin_amdgcn_sched_barrier(0);
    v16h ap[2];
#pragma unroll
    for (int kk = 0; kk < 2; ++kk) ap[kk] = load_a(pbuf + kk * 32, LDP, lane);

    // O += P @ V  (V tile is d-major in LDS)
#pragma unroll
    for (int t = 0; t < 4; ++t) {
#pragma unroll
      for (int kk = 0; kk < 2; ++kk) {
        v16h b = load_b(Vb + (t * 16) * LDP + kk * 32, LDP, lane);
        o[t] = wmma16(ap[kk], b, o[t]);
      }
    }
    __syncthreads();    // everyone done with buf 'cur' before it is re-staged
  }

  // normalize and store windowed output (f16)
  _Float16* O = Ow + ((size_t)(g * NH + head) * LWIN + q0) * DH;
  float inv[8];
#pragma unroll
  for (int r = 0; r < 8; ++r) inv[r] = 1.0f / lrow[r];
#pragma unroll
  for (int t = 0; t < 4; ++t) {
    const int d = t * 16 + lr;
#pragma unroll
    for (int r = 0; r < 8; ++r)
      O[(grp * 8 + r) * DH + d] = (_Float16)(o[t][r] * inv[r]);
  }
}

// ---------------------------------------------------------------------------
// Output projection: out = O @ Wo^T + bo (f32 out). The window->token
// permutation is folded into the A-fragment base. Wave computes 32x64.
// grid(48,8) block(32,8).
// ---------------------------------------------------------------------------
__global__ __launch_bounds__(256)
void proj_kernel(const _Float16* __restrict__ Ow,
                 const _Float16* __restrict__ Woh,
                 const float* __restrict__ bo,
                 float* __restrict__ out) {
  const int lane = threadIdx.x;
  const int m0 = (blockIdx.x * 8 + threadIdx.y) * 32;
  const int n0 = blockIdx.y * 64;
  const int grp = lane >> 4, lr = lane & 15;

  const int hrow  = m0 / WMAP;
  const int rem   = m0 % WMAP;
  const int g     = rem >> 6;
  const int lbase = hrow * 64 + (rem & 63);

  v8f acc[2][4] = {};
#pragma unroll
  for (int ks = 0; ks < 16; ++ks) {
    const int k0 = ks * 32;
    const int head = k0 >> 6;        // k-step stays inside one head block
    const int d0 = k0 & 63;
    const _Float16* Ab =
        Ow + ((size_t)(g * NH + head) * LWIN + lbase) * DH + d0;
    v16h a0 = load_a(Ab, DH, lane);
    v16h a1 = load_a(Ab + 16 * DH, DH, lane);
#pragma unroll
    for (int t = 0; t < 4; ++t) {
      v16h b = load_b(Woh + (n0 + t * 16) * D + k0, D, lane);
      acc[0][t] = wmma16(a0, b, acc[0][t]);
      acc[1][t] = wmma16(a1, b, acc[1][t]);
    }
  }
#pragma unroll
  for (int mi = 0; mi < 2; ++mi) {
#pragma unroll
    for (int t = 0; t < 4; ++t) {
      const int c = n0 + t * 16 + lr;
      const float bias = bo[c];
#pragma unroll
      for (int r = 0; r < 8; ++r)
        out[(size_t)(m0 + mi * 16 + grp * 8 + r) * D + c] = acc[mi][t][r] + bias;
    }
  }
}

// ---------------------------------------------------------------------------
extern "C" void kernel_launch(void* const* d_in, const int* in_sizes, int n_in,
                              void* d_out, int out_size, void* d_ws, size_t ws_size,
                              hipStream_t stream) {
  const float* x  = (const float*)d_in[0];
  const float* Wq = (const float*)d_in[1];
  const float* Wk = (const float*)d_in[2];
  const float* Wv = (const float*)d_in[3];
  const float* Wo = (const float*)d_in[4];
  const float* bo = (const float*)d_in[5];
  float* out = (float*)d_out;

  // workspace carve-up (halves)
  _Float16* ws  = (_Float16*)d_ws;
  _Float16* xh  = ws;                            // 12288*512
  _Float16* Wqh = xh  + (size_t)N_TOK * D;
  _Float16* Wkh = Wqh + (size_t)D * D;
  _Float16* Wvh = Wkh + (size_t)D * D;
  _Float16* Woh = Wvh + (size_t)D * D;
  _Float16* Qw  = Woh + (size_t)D * D;           // [6][8][2048][64]
  _Float16* Kw  = Qw  + (size_t)N_TOK * D;
  _Float16* Vt  = Kw  + (size_t)N_TOK * D;       // [6][8][64][2048]
  _Float16* Ow  = Vt  + (size_t)N_TOK * D;

  const int nx8 = N_TOK * D / 8;
  const int nw8 = D * D / 8;
  cvt_kernel<<<dim3((nx8 + 255) / 256), dim3(256), 0, stream>>>(x,  xh,  nx8);
  cvt_kernel<<<dim3((nw8 + 255) / 256), dim3(256), 0, stream>>>(Wq, Wqh, nw8);
  cvt_kernel<<<dim3((nw8 + 255) / 256), dim3(256), 0, stream>>>(Wk, Wkh, nw8);
  cvt_kernel<<<dim3((nw8 + 255) / 256), dim3(256), 0, stream>>>(Wv, Wvh, nw8);
  cvt_kernel<<<dim3((nw8 + 255) / 256), dim3(256), 0, stream>>>(Wo, Woh, nw8);

  qkv_kernel<<<dim3(48, 8, 3), dim3(32, 8), 0, stream>>>(xh, Wqh, Wkh, Wvh, Qw, Kw, Vt);
  attn_kernel<<<dim3(16, 8, 6), dim3(32, 8), 0, stream>>>(Qw, Kw, Vt, Ow);
  proj_kernel<<<dim3(48, 8), dim3(32, 8), 0, stream>>>(Ow, Woh, bo, out);
}
